// LlamaQuantizedMLP_43035572306191
// MI455X (gfx1250) — compile-verified, matrix operand order checked by reading, not run on hardware
//
#include <hip/hip_runtime.h>
#include <math.h>
#include <stdint.h>
#include <stddef.h>

// ---------------------------------------------------------------------------
// LlamaQuantizedMLP on MI455X (gfx1250):
//   fake-quant int8 (per-token acts, per-out-channel weights) => the matmuls
//   are EXACT int8 GEMMs with a rank-1 scale. We run them on
//   V_WMMA_I32_16X16X64_IU8 with register-blocked wave tiles:
//     gate/up : 64x32 per wave (MT=4, NT=2), A shared by gate & up
//               => 16 WMMA per 8 fragment loads (2 WMMA/load).
//     down    : 64x64 per wave (MT=4, NT=4)
//               => 16 WMMA per 8 fragment loads (2 WMMA/load).
//   __launch_bounds__(256, 1) lifts the default ~128-VGPR occupancy cap so
//   the 16 accumulator chains stay in registers (no scratch spills).
// ---------------------------------------------------------------------------

typedef __attribute__((ext_vector_type(8))) int v8i;

#define QMAX 127.0f

static constexpr int HIDDEN = 4096;
static constexpr int INTER  = 11008;
static constexpr int TOKENS = 2 * 2048;   // B * S

// ---------------------------------------------------------------------------
// Per-row symmetric int8 quantization (row-major [rows, cols]).
// Matches jnp: s = max(absmax/127, 1e-8); q = clip(rint(v/s), -128, 127).
// One 256-thread block per row; float4 / char4 vectorized (cols % 4 == 0).
// ---------------------------------------------------------------------------
__global__ void quant_rows_kernel(const float* __restrict__ in,
                                  int8_t* __restrict__ qout,
                                  float* __restrict__ scale,
                                  int cols) {
  const int row   = blockIdx.x;
  const int cols4 = cols >> 2;
  const float4* __restrict__ r4 = (const float4*)(in + (size_t)row * cols);

  __shared__ float red[256];
  float m = 0.0f;
  for (int c = threadIdx.x; c < cols4; c += 256) {
    const float4 v = r4[c];
    m = fmaxf(m, fmaxf(fmaxf(fabsf(v.x), fabsf(v.y)),
                       fmaxf(fabsf(v.z), fabsf(v.w))));
  }
  red[threadIdx.x] = m;
  __syncthreads();
#pragma unroll
  for (int s = 128; s > 0; s >>= 1) {
    if (threadIdx.x < s)
      red[threadIdx.x] = fmaxf(red[threadIdx.x], red[threadIdx.x + s]);
    __syncthreads();
  }
  const float sc = fmaxf(red[0] * (1.0f / QMAX), 1e-8f);
  if (threadIdx.x == 0) scale[row] = sc;
  const float inv = 1.0f / sc;

  char4* __restrict__ q4 = (char4*)(qout + (size_t)row * cols);
  for (int c = threadIdx.x; c < cols4; c += 256) {
    const float4 v = r4[c];
    char4 q;
    q.x = (int8_t)fminf(fmaxf(rintf(v.x * inv), -128.0f), 127.0f);
    q.y = (int8_t)fminf(fmaxf(rintf(v.y * inv), -128.0f), 127.0f);
    q.z = (int8_t)fminf(fmaxf(rintf(v.z * inv), -128.0f), 127.0f);
    q.w = (int8_t)fminf(fmaxf(rintf(v.w * inv), -128.0f), 127.0f);
    q4[c] = q;
  }
}

// ---------------------------------------------------------------------------
// Load one 16x64 int8 fragment (8 VGPRs / lane) per the CDNA5 8-bit A layout:
//   lanes 0-15  : row M = lane,    K bytes {0-7, 16-23, 32-39, 48-55}
//   lanes 16-31 : row M = lane-16, K bytes {8-15, 24-31, 40-47, 56-63}
// Rows are contiguous (row-major, stride ldk), so this is 4 aligned b64 loads.
// ---------------------------------------------------------------------------
__device__ __forceinline__ v8i load_frag8(const int8_t* __restrict__ base,
                                          int ldk, int row, int k0, int lane) {
  const int g = (lane >> 4) << 3;                       // 0 or 8 bytes
  const int8_t* __restrict__ p = base + (size_t)row * ldk + k0 + g;
  const uint2 q0 = *(const uint2*)(p);
  const uint2 q1 = *(const uint2*)(p + 16);
  const uint2 q2 = *(const uint2*)(p + 32);
  const uint2 q3 = *(const uint2*)(p + 48);
  v8i a;
  a[0] = (int)q0.x; a[1] = (int)q0.y;
  a[2] = (int)q1.x; a[3] = (int)q1.y;
  a[4] = (int)q2.x; a[5] = (int)q2.y;
  a[6] = (int)q3.x; a[7] = (int)q3.y;
  return a;
}

// ---------------------------------------------------------------------------
// Fused gate/up projection + SwiGLU, register-blocked 64x32 per wave:
//   h[t,n] = silu(accg * sx[t]*swg[n]) * (accu * sx[t]*swu[n])
// ---------------------------------------------------------------------------
__global__ void __launch_bounds__(256, 1)
gemm_gate_up_silu_kernel(const int8_t* __restrict__ xq,
                         const float* __restrict__ sx,
                         const int8_t* __restrict__ wgq,
                         const float* __restrict__ swg,
                         const int8_t* __restrict__ wuq,
                         const float* __restrict__ swu,
                         float* __restrict__ h,
                         int M, int N, int K) {
  constexpr int MT = 4, NT = 2;
  const int lane   = threadIdx.x & 31;
  const int wave   = blockIdx.x * (blockDim.x >> 5) + (threadIdx.x >> 5);
  const int tilesN = N / (16 * NT);
  const int tm     = (wave / tilesN) * (16 * MT);   // wave-uniform => EXEC full
  const int tn     = (wave % tilesN) * (16 * NT);
  if (tm >= M) return;

  const int fr = lane & 15;      // fragment row within a 16-row tile

  v8i accg[MT][NT] = {};
  v8i accu[MT][NT] = {};

  for (int k0 = 0; k0 < K; k0 += 64) {
    v8i a[MT];
#pragma unroll
    for (int i = 0; i < MT; ++i)
      a[i] = load_frag8(xq, K, tm + 16 * i + fr, k0, lane);
    v8i bg[NT], bu[NT];
#pragma unroll
    for (int j = 0; j < NT; ++j) {
      bg[j] = load_frag8(wgq, K, tn + 16 * j + fr, k0, lane);
      bu[j] = load_frag8(wuq, K, tn + 16 * j + fr, k0, lane);
    }
#pragma unroll
    for (int i = 0; i < MT; ++i) {
#pragma unroll
      for (int j = 0; j < NT; ++j) {
        accg[i][j] = __builtin_amdgcn_wmma_i32_16x16x64_iu8(
            true, a[i], true, bg[j], accg[i][j], false, false);
        accu[i][j] = __builtin_amdgcn_wmma_i32_16x16x64_iu8(
            true, a[i], true, bu[j], accu[i][j], false, false);
      }
    }
  }

  // D layout: elem r of lane -> M = tile_m + r + 8*(lane>>4), N = tile_n + (lane&15)
#pragma unroll
  for (int j = 0; j < NT; ++j) {
    const int n  = tn + 16 * j + (lane & 15);
    const float sg = swg[n];
    const float su = swu[n];
#pragma unroll
    for (int i = 0; i < MT; ++i) {
      const int mb = tm + 16 * i + ((lane >> 4) << 3);
#pragma unroll
      for (int r = 0; r < 8; ++r) {
        const int m    = mb + r;
        const float sm = sx[m];
        const float gate = (float)accg[i][j][r] * sm * sg;
        const float up   = (float)accu[i][j][r] * sm * su;
        const float sig  = 1.0f / (1.0f + __expf(-gate));
        h[(size_t)m * N + n] = gate * sig * up;
      }
    }
  }
}

// ---------------------------------------------------------------------------
// Down projection, register-blocked 64x64 per wave:
//   out[t,n] = acc * sh[t] * swd[n]
// ---------------------------------------------------------------------------
__global__ void __launch_bounds__(256, 1)
gemm_down_kernel(const int8_t* __restrict__ hq,
                 const float* __restrict__ sh,
                 const int8_t* __restrict__ wdq,
                 const float* __restrict__ swd,
                 float* __restrict__ out,
                 int M, int N, int K) {
  constexpr int MT = 4, NT = 4;
  const int lane   = threadIdx.x & 31;
  const int wave   = blockIdx.x * (blockDim.x >> 5) + (threadIdx.x >> 5);
  const int tilesN = N / (16 * NT);
  const int tm     = (wave / tilesN) * (16 * MT);
  const int tn     = (wave % tilesN) * (16 * NT);
  if (tm >= M) return;

  const int fr = lane & 15;

  v8i acc[MT][NT] = {};

  for (int k0 = 0; k0 < K; k0 += 64) {
    v8i a[MT];
#pragma unroll
    for (int i = 0; i < MT; ++i)
      a[i] = load_frag8(hq, K, tm + 16 * i + fr, k0, lane);
    v8i b[NT];
#pragma unroll
    for (int j = 0; j < NT; ++j)
      b[j] = load_frag8(wdq, K, tn + 16 * j + fr, k0, lane);
#pragma unroll
    for (int i = 0; i < MT; ++i) {
#pragma unroll
      for (int j = 0; j < NT; ++j) {
        acc[i][j] = __builtin_amdgcn_wmma_i32_16x16x64_iu8(
            true, a[i], true, b[j], acc[i][j], false, false);
      }
    }
  }

#pragma unroll
  for (int j = 0; j < NT; ++j) {
    const int n  = tn + 16 * j + (lane & 15);
    const float sn = swd[n];
#pragma unroll
    for (int i = 0; i < MT; ++i) {
      const int mb = tm + 16 * i + ((lane >> 4) << 3);
#pragma unroll
      for (int r = 0; r < 8; ++r) {
        const int m = mb + r;
        out[(size_t)m * N + n] = (float)acc[i][j][r] * sh[m] * sn;
      }
    }
  }
}

// ---------------------------------------------------------------------------
// Launcher
// ---------------------------------------------------------------------------
static inline size_t align_up(size_t x, size_t a) { return (x + a - 1) & ~(a - 1); }

extern "C" void kernel_launch(void* const* d_in, const int* in_sizes, int n_in,
                              void* d_out, int out_size, void* d_ws, size_t ws_size,
                              hipStream_t stream) {
  (void)in_sizes; (void)n_in; (void)out_size; (void)ws_size;

  const float* x      = (const float*)d_in[0];   // [TOKENS, HIDDEN]
  const float* w_gate = (const float*)d_in[1];   // [INTER, HIDDEN]
  const float* w_up   = (const float*)d_in[2];   // [INTER, HIDDEN]
  const float* w_down = (const float*)d_in[3];   // [HIDDEN, INTER]
  float* out          = (float*)d_out;           // [TOKENS, HIDDEN]

  // ---- workspace carve-up ----
  uint8_t* ws = (uint8_t*)d_ws;
  size_t off = 0;
  auto take = [&](size_t bytes) -> uint8_t* {
    uint8_t* p = ws + off;
    off = align_up(off + bytes, 256);
    return p;
  };
  int8_t* xq  = (int8_t*)take((size_t)TOKENS * HIDDEN);
  int8_t* wgq = (int8_t*)take((size_t)INTER  * HIDDEN);
  int8_t* wuq = (int8_t*)take((size_t)INTER  * HIDDEN);
  int8_t* wdq = (int8_t*)take((size_t)HIDDEN * INTER);
  int8_t* hq  = (int8_t*)take((size_t)TOKENS * INTER);
  float*  h   = (float*) take((size_t)TOKENS * INTER * sizeof(float));
  float*  sx  = (float*) take((size_t)TOKENS * sizeof(float));
  float*  swg = (float*) take((size_t)INTER  * sizeof(float));
  float*  swu = (float*) take((size_t)INTER  * sizeof(float));
  float*  swd = (float*) take((size_t)HIDDEN * sizeof(float));
  float*  sh  = (float*) take((size_t)TOKENS * sizeof(float));

  // ---- 1) quantize activations + weights (per-row int8) ----
  quant_rows_kernel<<<TOKENS, 256, 0, stream>>>(x,      xq,  sx,  HIDDEN);
  quant_rows_kernel<<<INTER,  256, 0, stream>>>(w_gate, wgq, swg, HIDDEN);
  quant_rows_kernel<<<INTER,  256, 0, stream>>>(w_up,   wuq, swu, HIDDEN);
  quant_rows_kernel<<<HIDDEN, 256, 0, stream>>>(w_down, wdq, swd, INTER);

  // ---- 2) gate/up projections + SwiGLU (iu8 WMMA, 64x32 per wave) ----
  {
    const int waves  = (TOKENS / 64) * (INTER / 32);   // 64 * 344
    const int wavesB = 256 / 32;                       // 8 waves / block
    const int blocks = (waves + wavesB - 1) / wavesB;
    gemm_gate_up_silu_kernel<<<blocks, 256, 0, stream>>>(
        xq, sx, wgq, swg, wuq, swu, h, TOKENS, INTER, HIDDEN);
  }

  // ---- 3) quantize h per token ----
  quant_rows_kernel<<<TOKENS, 256, 0, stream>>>(h, hq, sh, INTER);

  // ---- 4) down projection (iu8 WMMA, 64x64 per wave) ----
  {
    const int waves  = (TOKENS / 64) * (HIDDEN / 64);  // 64 * 64
    const int wavesB = 256 / 32;
    const int blocks = (waves + wavesB - 1) / wavesB;
    gemm_down_kernel<<<blocks, 256, 0, stream>>>(
        hq, sh, wdq, swd, out, TOKENS, HIDDEN, INTER);
  }
}
